// ToyDenSaE_24927990186281
// MI455X (gfx1250) — matrix operand order repaired and checked by reading, not compiled, not run on hardware
//
#include <hip/hip_runtime.h>
#include <hip/hip_bf16.h>
#include <stddef.h>

// ---------------------------------------------------------------------------
// ISTA sparse-coding loop on MI455X (gfx1250), bf16 WMMA w/ f32 accumulation.
//   x:[4096,1024]  A:[512,1024]  B:[8192,1024]
//   10 iters:  res = x - z@A - s@B
//              z  += 0.2*(res@A^T)
//              s   = relu(s + 0.2*(res@B^T) - 0.1)
// GEMM scheme: block = 128 threads (4 waves), tile 128(M) x 64(N).
//   - B k-slab (64 rows x 32 K bf16) staged in LDS via async global->LDS,
//     double buffered, shared by all 4 waves (4x traffic reduction).
//   - Each wave: 2 M-frags x 4 N-frags = 8 WMMA per 32-K step.
// ---------------------------------------------------------------------------

#define USE_ASYNC_LDS 1

#define M_BATCH 4096
#define D_IN    1024
#define D_Z     512
#define D_S     8192
#define N_ITERS 10
#define ALPHA   0.2f
#define THRESH  0.1f

typedef __attribute__((ext_vector_type(16))) __bf16 v16bf;
typedef __attribute__((ext_vector_type(8)))  __bf16 v8bf;
typedef __attribute__((ext_vector_type(8)))  float  v8f;

union Frag16 { v16bf v; v8bf h[2]; };

__device__ __forceinline__ unsigned short f32_to_bf16(float f) {
  union { float f; unsigned u; } c; c.f = f;
  unsigned r = c.u + 0x7FFFu + ((c.u >> 16) & 1u);   // round-to-nearest-even
  return (unsigned short)(r >> 16);
}

// C/D layout: VGPR r -> M = r (lanes 0-15) or 8+r (lanes 16-31), N = lane&15.
__device__ __forceinline__ int cd_row(int m0, int r, int lane) {
  return m0 + ((lane < 16) ? r : 8 + r);
}

// ---------------------------------------------------------------------------
// Block-cooperative GEMM inner loop with LDS-staged B operand.
//   acc[mi][j] += A(m0+16mi ..)[K] x Bt(n0+16j ..)^T[K]
//   A  : row-major [*, lda] bf16 (per-lane row; split-K A-operand layout)
//   Bt : row-major transposed operand [*, ldb] bf16 (row n = col n of K x N)
// LDS slab layout: row n (0..63) holds 64B of K-chunk, stride 80B (40 halves)
// to spread ds_load_b128 starting banks.
// ---------------------------------------------------------------------------
template <int MF, int NF>
__device__ __forceinline__ void gemm_lds(v8f (&acc)[MF][NF],
                                         const unsigned short* __restrict__ Au,
                                         int lda, int m0,
                                         const unsigned short* __restrict__ Btu,
                                         int ldb, int n0, int K) {
  static_assert(NF == 4, "block N-tile fixed at 64");
  const int tid  = threadIdx.x;          // 0..127
  const int lane = tid & 31;

  __shared__ alignas(16) unsigned short ldsB[2][NF * 16 * 40]; // 2 x 5120 B

  // --- async fill of one 64x32 bf16 slab: 256 b128 pieces, 2 per thread ---
  auto fill = [&](int buf, int k) {
#pragma unroll
    for (int u = 0; u < 2; ++u) {
      const int c = tid * 2 + u;         // piece id 0..255
      const int n = c >> 2;              // slab row 0..63
      const int p = c & 3;               // 16B piece within row
      const unsigned short* g = Btu + (size_t)(n0 + n) * ldb + (k + p * 8);
#if USE_ASYNC_LDS
      unsigned loff = (unsigned)(size_t)&ldsB[buf][n * 40 + p * 8];
      asm volatile("global_load_async_to_lds_b128 %0, %1, off"
                   :: "v"(loff), "v"(g) : "memory");
#else
      *(uint4*)&ldsB[buf][n * 40 + p * 8] = *(const uint4*)g;
#endif
    }
  };

  // A-operand row pointers: lanes 0-15 K {0..7|16..23}, lanes 16-31 {8..15|24..31}
  const unsigned short* ap[MF];
#pragma unroll
  for (int mi = 0; mi < MF; ++mi)
    ap[mi] = Au + (size_t)(m0 + mi * 16 + (lane & 15)) * lda +
             ((lane < 16) ? 0 : 8);

  const int nsteps = K / 32;
  fill(0, 0);
  for (int i = 0; i < nsteps; ++i) {
    const int k = i * 32;
    if (i + 1 < nsteps) {
      fill((i + 1) & 1, k + 32);
#if USE_ASYNC_LDS
      __builtin_amdgcn_s_wait_asynccnt(2);   // previous slab complete (in-order)
#endif
    } else {
#if USE_ASYNC_LDS
      __builtin_amdgcn_s_wait_asynccnt(0);
#endif
    }
    __syncthreads();                         // slab visible to all waves

    Frag16 af[MF];
#pragma unroll
    for (int mi = 0; mi < MF; ++mi) {
      af[mi].h[0] = *(const v8bf*)(ap[mi] + k);
      af[mi].h[1] = *(const v8bf*)(ap[mi] + k + 16);
      __builtin_prefetch((const void*)(ap[mi] + k + 128), 0, 1);
    }
#pragma unroll
    for (int j = 0; j < NF; ++j) {
      // B-operand: lanes 0-15 hold K {0..7|16..23}, lanes 16-31 {8..15|24..31}
      const unsigned short* fb =
          &ldsB[i & 1][(j * 16 + (lane & 15)) * 40 + ((lane < 16) ? 0 : 8)];
      Frag16 bf;
      bf.h[0] = *(const v8bf*)(fb);
      bf.h[1] = *(const v8bf*)(fb + 16);
#pragma unroll
      for (int mi = 0; mi < MF; ++mi)
        acc[mi][j] = __builtin_amdgcn_wmma_f32_16x16x32_bf16(
            false, af[mi].v, false, bf.v, (short)0, acc[mi][j], false, false);
    }
    __syncthreads();                         // done reading before overwrite
  }
}

// ---------------------------------------------------------------------------
// Elementwise / preprocessing kernels
// ---------------------------------------------------------------------------
__global__ void k_zero16(uint4* __restrict__ p, int n16) {
  int i = blockIdx.x * blockDim.x + threadIdx.x;
  if (i < n16) { uint4 z = {0u, 0u, 0u, 0u}; p[i] = z; }
}

__global__ void k_convert_bf16(unsigned short* __restrict__ dst,
                               const float* __restrict__ src, int n) {
  int i = blockIdx.x * blockDim.x + threadIdx.x;
  if (i < n) dst[i] = f32_to_bf16(src[i]);
}

// dst[c*rows + r] = bf16(src[r*cols + c]); 32x32 LDS tile, block 32x8
__global__ void k_transpose_bf16(unsigned short* __restrict__ dst,
                                 const float* __restrict__ src,
                                 int rows, int cols) {
  __shared__ unsigned short tile[32][33];
  const int c0 = blockIdx.x * 32, r0 = blockIdx.y * 32;
  const int tx = threadIdx.x, ty = threadIdx.y;
#pragma unroll
  for (int i = 0; i < 32; i += 8)
    tile[ty + i][tx] = f32_to_bf16(src[(size_t)(r0 + ty + i) * cols + (c0 + tx)]);
  __syncthreads();
#pragma unroll
  for (int i = 0; i < 32; i += 8)
    dst[(size_t)(c0 + ty + i) * rows + (r0 + tx)] = tile[tx][ty + i];
}

// ---------------------------------------------------------------------------
// GEMM kernels: 128 threads = 4 waves; block tile 128(M) x 64(N);
// wave w owns rows [blk*128 + 32w, +32), 2 M-frags x 4 N-frags.
// ---------------------------------------------------------------------------
__global__ __launch_bounds__(128)
void k_residual(const float* __restrict__ x,
                const unsigned short* __restrict__ zbf,
                const unsigned short* __restrict__ Atbf,
                const unsigned short* __restrict__ sbf,
                const unsigned short* __restrict__ Btbf,
                unsigned short* __restrict__ resbf) {
  const int lane = threadIdx.x & 31, wave = threadIdx.x >> 5;
  const int m0 = blockIdx.x * 128 + wave * 32;
  const int n0 = blockIdx.y * 64;
  v8f acc[2][4] = {};
  gemm_lds<2, 4>(acc, zbf, D_Z, m0, Atbf, D_Z, n0, D_Z);     // z @ A
  gemm_lds<2, 4>(acc, sbf, D_S, m0, Btbf, D_S, n0, D_S);     // s @ B
#pragma unroll
  for (int mi = 0; mi < 2; ++mi)
#pragma unroll
    for (int j = 0; j < 4; ++j)
#pragma unroll
      for (int r = 0; r < 8; ++r) {
        const int row = cd_row(m0 + mi * 16, r, lane);
        const int col = n0 + j * 16 + (lane & 15);
        const size_t idx = (size_t)row * D_IN + col;
        resbf[idx] = f32_to_bf16(x[idx] - acc[mi][j][r]);
      }
}

__global__ __launch_bounds__(128)
void k_update_z(const unsigned short* __restrict__ resbf,
                const unsigned short* __restrict__ Abf,
                float* __restrict__ z, unsigned short* __restrict__ zbf) {
  const int lane = threadIdx.x & 31, wave = threadIdx.x >> 5;
  const int m0 = blockIdx.x * 128 + wave * 32;
  const int n0 = blockIdx.y * 64;
  v8f acc[2][4] = {};
  gemm_lds<2, 4>(acc, resbf, D_IN, m0, Abf, D_IN, n0, D_IN); // res @ A^T
#pragma unroll
  for (int mi = 0; mi < 2; ++mi)
#pragma unroll
    for (int j = 0; j < 4; ++j)
#pragma unroll
      for (int r = 0; r < 8; ++r) {
        const int row = cd_row(m0 + mi * 16, r, lane);
        const int col = n0 + j * 16 + (lane & 15);
        const size_t idx = (size_t)row * D_Z + col;
        const float v = z[idx] + ALPHA * acc[mi][j][r];
        z[idx] = v;
        zbf[idx] = f32_to_bf16(v);
      }
}

__global__ __launch_bounds__(128)
void k_update_s(const unsigned short* __restrict__ resbf,
                const unsigned short* __restrict__ Bbf,
                float* __restrict__ s, unsigned short* __restrict__ sbf) {
  const int lane = threadIdx.x & 31, wave = threadIdx.x >> 5;
  const int m0 = blockIdx.x * 128 + wave * 32;
  const int n0 = blockIdx.y * 64;
  v8f acc[2][4] = {};
  gemm_lds<2, 4>(acc, resbf, D_IN, m0, Bbf, D_IN, n0, D_IN); // res @ B^T
#pragma unroll
  for (int mi = 0; mi < 2; ++mi)
#pragma unroll
    for (int j = 0; j < 4; ++j)
#pragma unroll
      for (int r = 0; r < 8; ++r) {
        const int row = cd_row(m0 + mi * 16, r, lane);
        const int col = n0 + j * 16 + (lane & 15);
        const size_t idx = (size_t)row * D_S + col;
        float v = s[idx] + ALPHA * acc[mi][j][r] - THRESH; // one-sided shrink
        v = fmaxf(v, 0.0f);
        s[idx] = v;
        sbf[idx] = f32_to_bf16(v);
      }
}

__global__ __launch_bounds__(128)
void k_reconstruct(const unsigned short* __restrict__ zbf,
                   const unsigned short* __restrict__ Atbf,
                   const unsigned short* __restrict__ sbf,
                   const unsigned short* __restrict__ Btbf,
                   float* __restrict__ xhat, float* __restrict__ xdense,
                   float* __restrict__ xsparse) {
  const int lane = threadIdx.x & 31, wave = threadIdx.x >> 5;
  const int m0 = blockIdx.x * 128 + wave * 32;
  const int n0 = blockIdx.y * 64;
  v8f accD[2][4] = {}, accS[2][4] = {};
  gemm_lds<2, 4>(accD, zbf, D_Z, m0, Atbf, D_Z, n0, D_Z);    // x_dense
  gemm_lds<2, 4>(accS, sbf, D_S, m0, Btbf, D_S, n0, D_S);    // x_sparse
#pragma unroll
  for (int mi = 0; mi < 2; ++mi)
#pragma unroll
    for (int j = 0; j < 4; ++j)
#pragma unroll
      for (int r = 0; r < 8; ++r) {
        const int row = cd_row(m0 + mi * 16, r, lane);
        const int col = n0 + j * 16 + (lane & 15);
        const size_t idx = (size_t)row * D_IN + col;
        const float d = accD[mi][j][r], sp = accS[mi][j][r];
        xdense[idx]  = d;
        xsparse[idx] = sp;
        xhat[idx]    = d + sp;
      }
}

// ---------------------------------------------------------------------------
extern "C" void kernel_launch(void* const* d_in, const int* in_sizes, int n_in,
                              void* d_out, int out_size, void* d_ws, size_t ws_size,
                              hipStream_t stream) {
  (void)in_sizes; (void)n_in; (void)out_size; (void)ws_size;
  const float* x = (const float*)d_in[0];
  const float* A = (const float*)d_in[1];
  const float* B = (const float*)d_in[2];

  // Output layout (f32, return order): x_hat, z, s, x_dense, x_sparse
  float* out     = (float*)d_out;
  float* xhat    = out;
  float* z       = xhat + (size_t)M_BATCH * D_IN;
  float* s       = z    + (size_t)M_BATCH * D_Z;
  float* xdense  = s    + (size_t)M_BATCH * D_S;
  float* xsparse = xdense + (size_t)M_BATCH * D_IN;

  // Workspace layout (bf16 shadows; ~110 MB)
  char* ws = (char*)d_ws;
  unsigned short* Abf  = (unsigned short*)(ws);
  unsigned short* Atbf = Abf  + (size_t)D_Z  * D_IN;
  unsigned short* Bbf  = Atbf + (size_t)D_IN * D_Z;
  unsigned short* Btbf = Bbf  + (size_t)D_S  * D_IN;
  unsigned short* zbf  = Btbf + (size_t)D_IN * D_S;
  unsigned short* sbf  = zbf  + (size_t)M_BATCH * D_Z;
  unsigned short* resbf= sbf  + (size_t)M_BATCH * D_S;

  // --- Preprocessing: bf16 copies of A, B and their transposes ---
  {
    int nA = D_Z * D_IN, nB = D_S * D_IN;
    k_convert_bf16<<<(nA + 255) / 256, 256, 0, stream>>>(Abf, A, nA);
    k_convert_bf16<<<(nB + 255) / 256, 256, 0, stream>>>(Bbf, B, nB);
    k_transpose_bf16<<<dim3(D_IN / 32, D_Z / 32), dim3(32, 8), 0, stream>>>(
        Atbf, A, D_Z, D_IN);
    k_transpose_bf16<<<dim3(D_IN / 32, D_S / 32), dim3(32, 8), 0, stream>>>(
        Btbf, B, D_S, D_IN);
  }
  // --- Zero init: z|s (contiguous f32 in d_out) and zbf|sbf (contiguous) ---
  {
    int n16a = (int)(((size_t)M_BATCH * (D_Z + D_S) * sizeof(float)) / 16);
    k_zero16<<<(n16a + 255) / 256, 256, 0, stream>>>((uint4*)z, n16a);
    int n16b = (int)(((size_t)M_BATCH * (D_Z + D_S) * sizeof(unsigned short)) / 16);
    k_zero16<<<(n16b + 255) / 256, 256, 0, stream>>>((uint4*)zbf, n16b);
  }

  const dim3 blk(128);
  const dim3 gRes(M_BATCH / 128, D_IN / 64);   // 32 x 16
  const dim3 gZ(M_BATCH / 128, D_Z / 64);      // 32 x 8
  const dim3 gS(M_BATCH / 128, D_S / 64);      // 32 x 128

  for (int it = 0; it < N_ITERS; ++it) {
    k_residual<<<gRes, blk, 0, stream>>>(x, zbf, Atbf, sbf, Btbf, resbf);
    k_update_z<<<gZ, blk, 0, stream>>>(resbf, Abf, z, zbf);
    k_update_s<<<gS, blk, 0, stream>>>(resbf, Bbf, s, sbf);
  }
  k_reconstruct<<<gRes, blk, 0, stream>>>(zbf, Atbf, sbf, Btbf,
                                          xhat, xdense, xsparse);
}
